// RotatedLoRALinear_6279242187444
// MI455X (gfx1250) — compile-verified, matrix operand order checked by reading, not run on hardware
//
#include <hip/hip_runtime.h>
#include <stdint.h>

typedef __attribute__((ext_vector_type(16))) _Float16 v16h;
typedef __attribute__((ext_vector_type(8)))  _Float16 v8h;
typedef __attribute__((ext_vector_type(8)))  float    v8f;
typedef __attribute__((ext_vector_type(4)))  int      v4i;

#define DDIM 4096
#define KP   4160          // D + 16 LoRA columns + 48 zero pad (multiple of BK=64)
#define ROWS 8192
#define SCALING 0.01f

#define BM 128
#define BN 128
#define BK 64
#define LDB 72             // 64 halves + 8 pad (144B rows -> conflict-free frag reads)

#if defined(__has_builtin)
#if __has_builtin(__builtin_amdgcn_global_load_async_to_lds_b128) && __has_builtin(__builtin_amdgcn_s_wait_asynccnt)
#define USE_ASYNC 1
#endif
#endif
#ifndef USE_ASYNC
#define USE_ASYNC 0
#endif

#if USE_ASYNC
typedef __attribute__((address_space(1))) v4i as1_v4i;   // global b128 payload
typedef __attribute__((address_space(3))) v4i as3_v4i;   // LDS b128 payload
#endif

// ---------------------------------------------------------------------------
// Kernel 1: per row -- FWHT (12 stages in LDS) -> f16 A operand cols [0,4096)
//           t = x @ lora_A (rank 16) -> cols [4096,4112), zeros to 4160
// ---------------------------------------------------------------------------
__global__ __launch_bounds__(256) void fwht_lora_kernel(
    const float* __restrict__ x, const float* __restrict__ loraA,
    _Float16* __restrict__ gA) {
  __shared__ float sx[DDIM];       // 16 KB
  __shared__ float st[16 * 256];   // 16 KB  (layout [k][256])
  const int row = blockIdx.x;
  const int tid = threadIdx.x;
  const float* xr = x + (size_t)row * DDIM;

  float tacc[16];
#pragma unroll
  for (int k = 0; k < 16; ++k) tacc[k] = 0.f;

  for (int j = tid; j < DDIM; j += 256) {
    float v = xr[j];
    sx[j] = v;
    const float* ar = loraA + (size_t)j * 16;
#pragma unroll
    for (int k = 0; k < 16; ++k) tacc[k] += v * ar[k];
  }
#pragma unroll
  for (int k = 0; k < 16; ++k) st[k * 256 + tid] = tacc[k];
  __syncthreads();

  // 12 butterfly stages
  for (int lh = 0; lh < 12; ++lh) {
    const int h = 1 << lh;
    for (int idx = tid; idx < DDIM / 2; idx += 256) {
      const int pos = idx & (h - 1);
      const int i0 = ((idx ^ pos) << 1) | pos;
      const int i1 = i0 + h;
      const float a = sx[i0], b = sx[i1];
      sx[i0] = a + b;
      sx[i1] = a - b;
    }
    __syncthreads();
  }

  // write normalized FWHT result as f16
  _Float16* ga = gA + (size_t)row * KP;
  for (int j = tid; j < DDIM; j += 256)
    ga[j] = (_Float16)(sx[j] * 0.015625f);   // 1/sqrt(4096)

  // reduce t across the 256 threads
  for (int s = 128; s > 0; s >>= 1) {
    if (tid < s) {
#pragma unroll
      for (int k = 0; k < 16; ++k) st[k * 256 + tid] += st[k * 256 + tid + s];
    }
    __syncthreads();
  }
  if (tid < KP - DDIM)   // 64 tail columns: 16 LoRA values, rest zero
    ga[DDIM + tid] = (tid < 16) ? (_Float16)st[tid * 256] : (_Float16)0.0f;
}

// ---------------------------------------------------------------------------
// Kernel 2: B operand rows: n-th row = [ W[n,:] | 0.01*lora_B[0..15,n] | 0 ]
// ---------------------------------------------------------------------------
__global__ __launch_bounds__(256) void prep_b_kernel(
    const float* __restrict__ W, const float* __restrict__ loraB,
    _Float16* __restrict__ gB) {
  const int n = blockIdx.x;
  const float* wr = W + (size_t)n * DDIM;
  _Float16* br = gB + (size_t)n * KP;
  for (int j = threadIdx.x; j < KP; j += 256) {
    float v;
    if (j < DDIM) v = wr[j];
    else {
      const int k = j - DDIM;
      v = (k < 16) ? SCALING * loraB[(size_t)k * DDIM + n] : 0.f;
    }
    br[j] = (_Float16)v;
  }
}

static __device__ inline v16h cat8(v8h lo, v8h hi) {
  return __builtin_shufflevector(lo, hi, 0, 1, 2, 3, 4, 5, 6, 7,
                                 8, 9, 10, 11, 12, 13, 14, 15);
}

// ---------------------------------------------------------------------------
// Kernel 3: C = A(8192xKP,f16) x B(4096xKP,f16)^T + bias
//           double-buffered async global->LDS, v_wmma_f32_16x16x32_f16 core
// ---------------------------------------------------------------------------
__global__ __launch_bounds__(256) void gemm_kernel(
    const _Float16* __restrict__ gA, const _Float16* __restrict__ gB,
    const float* __restrict__ bias, float* __restrict__ out) {
  __shared__ _Float16 sA[2][BM * LDB];   // 2 x 18 KB
  __shared__ _Float16 sB[2][BN * LDB];   // 2 x 18 KB

  const int tid   = threadIdx.x;
  const int tileM = blockIdx.x * BM;
  const int tileN = blockIdx.y * BN;
  const int wave  = tid >> 5;
  const int lane  = tid & 31;
  const int waveM = (wave & 1) * 64;   // 2x4 wave grid: 64x32 per wave
  const int waveN = (wave >> 1) * 32;
  const int m16   = lane & 15;
  const int koff  = (lane >> 4) * 8;   // ISA 16-bit A/B fragment layout

  v8f acc[4][2];
#pragma unroll
  for (int mi = 0; mi < 4; ++mi)
#pragma unroll
    for (int ni = 0; ni < 2; ++ni) acc[mi][ni] = (v8f)(0.f);

  // tile loader: 128 rows x 64 halves = 1024 16B chunks per operand tile
  auto load_tile = [&](int kt, int buf) {
    const int k0 = kt * BK;
#pragma unroll
    for (int c = 0; c < 4; ++c) {
      const int q  = tid + c * 256;
      const int r  = q >> 3;
      const int kh = (q & 7) * 8;       // half offset within the 64-wide row
      const _Float16* srcA = gA + (size_t)(tileM + r) * KP + k0 + kh;
      const _Float16* srcB = gB + (size_t)(tileN + r) * KP + k0 + kh;
#if USE_ASYNC
      __builtin_amdgcn_global_load_async_to_lds_b128(
          (as1_v4i*)srcA, (as3_v4i*)&sA[buf][r * LDB + kh], 0, 0);
      __builtin_amdgcn_global_load_async_to_lds_b128(
          (as1_v4i*)srcB, (as3_v4i*)&sB[buf][r * LDB + kh], 0, 0);
#else
      *(v8h*)&sA[buf][r * LDB + kh] = *(const v8h*)srcA;
      *(v8h*)&sB[buf][r * LDB + kh] = *(const v8h*)srcB;
#endif
    }
  };

  const int nK = KP / BK;   // 65
  load_tile(0, 0);          // prologue prefetch

  for (int kt = 0; kt < nK; ++kt) {
    const int buf = kt & 1;
#if USE_ASYNC
    __builtin_amdgcn_s_wait_asynccnt(0);   // my tile-kt data landed in LDS
#endif
    __syncthreads();                       // everyone's data visible; buf^1 free
    if (kt + 1 < nK) load_tile(kt + 1, buf ^ 1);   // prefetch overlaps compute

#pragma unroll
    for (int ks = 0; ks < 2; ++ks) {       // two K=32 sub-steps per tile
      const int kb = ks * 32;
      v16h af[4], bf[2];
#pragma unroll
      for (int mi = 0; mi < 4; ++mi) {
        const int rr = waveM + mi * 16 + m16;
        af[mi] = cat8(*(const v8h*)&sA[buf][rr * LDB + kb + koff],
                      *(const v8h*)&sA[buf][rr * LDB + kb + 16 + koff]);
      }
#pragma unroll
      for (int ni = 0; ni < 2; ++ni) {
        const int rr = waveN + ni * 16 + m16;
        bf[ni] = cat8(*(const v8h*)&sB[buf][rr * LDB + kb + koff],
                      *(const v8h*)&sB[buf][rr * LDB + kb + 16 + koff]);
      }
#pragma unroll
      for (int mi = 0; mi < 4; ++mi)
#pragma unroll
        for (int ni = 0; ni < 2; ++ni)
          acc[mi][ni] = __builtin_amdgcn_wmma_f32_16x16x32_f16(
              false, af[mi], false, bf[ni], (short)0, acc[mi][ni], false, false);
    }
  }

  // epilogue: + bias, store f32 (C layout: VGPR i -> M = 8*(lane>>4)+i, N = lane&15)
  const int mrow0 = 8 * (lane >> 4);
#pragma unroll
  for (int ni = 0; ni < 2; ++ni) {
    const int col = tileN + waveN + ni * 16 + m16;
    const float bb = bias[col];
#pragma unroll
    for (int mi = 0; mi < 4; ++mi) {
#pragma unroll
      for (int i = 0; i < 8; ++i) {
        const int rowg = tileM + waveM + mi * 16 + mrow0 + i;
        out[(size_t)rowg * DDIM + col] = acc[mi][ni][i] + bb;
      }
    }
  }
}

// ---------------------------------------------------------------------------
extern "C" void kernel_launch(void* const* d_in, const int* in_sizes, int n_in,
                              void* d_out, int out_size, void* d_ws, size_t ws_size,
                              hipStream_t stream) {
  const float* x  = (const float*)d_in[0];
  const float* W  = (const float*)d_in[1];
  const float* b  = (const float*)d_in[2];
  const float* lA = (const float*)d_in[3];
  const float* lB = (const float*)d_in[4];
  float* out = (float*)d_out;

  _Float16* gA = (_Float16*)d_ws;                                   // 8192*4160*2 B
  _Float16* gB = (_Float16*)((char*)d_ws + (size_t)ROWS * KP * 2);  // 4096*4160*2 B

  fwht_lora_kernel<<<ROWS, 256, 0, stream>>>(x, lA, gA);
  prep_b_kernel<<<DDIM, 256, 0, stream>>>(W, lB, gB);
  gemm_kernel<<<dim3(ROWS / BM, DDIM / BN), 256, 0, stream>>>(gA, gB, b, out);
}
